// ForecastPFN_TSMixer_3453153706387
// MI455X (gfx1250) — matrix operand order repaired and checked by reading, not compiled, not run on hardware
//
#include <hip/hip_runtime.h>
#include <hip/hip_bf16.h>
#include <math.h>

typedef __attribute__((ext_vector_type(16))) _Float16 v16h;
typedef __attribute__((ext_vector_type(8)))  _Float16 v8h;
typedef __attribute__((ext_vector_type(8)))  float    v8f;
typedef __attribute__((ext_vector_type(4)))  float    v4f;

#define T_HIST 100
#define SEQL   101
#define DM     72
#define XROWS  112           // padded residual rows
#define XS     80            // padded residual row stride (f32)
#define NWAVE  8

// LDS byte layout (all 16B aligned)
#define XRES_BYTES (XROWS*XS*4)                 // 35840
#define AB_BYTES   21504                        // 10752 f16 (covers 80x128 and 112x96)
#define SCR_FLOATS 192
#define SMEM_BYTES (XRES_BYTES + 3*AB_BYTES + SCR_FLOATS*4)   // 101120 B

// precomputed weight image layout in d_ws (f16, bias folded in at k==Kreal)
#define IMG_TW1_SZ (80*128)     // 10240
#define IMG_TW2_SZ (112*96)     // 10752
#define IMG_CW_SZ  (80*96)      // 7680
#define OFF_TW1_0  0
#define OFF_TW2_0  10240
#define OFF_CW1_0  20992
#define OFF_CW2_0  28672
#define OFF_TW1_1  36352
#define OFF_TW2_1  46592
#define OFF_CW1_1  57344
#define OFF_CW2_1  65024
#define IMG_TOTAL  72704
#define WS_NEEDED  ((size_t)IMG_TOTAL * 2)

__device__ __forceinline__ int iclampi(int v, int lo, int hi) {
    return v < lo ? lo : (v > hi ? hi : v);
}
__device__ __forceinline__ int imini(int a, int b) { return a < b ? a : b; }

// async copy global -> LDS in 16B chunks (ASYNCcnt-tracked), then wait.
__device__ __forceinline__ void async_stage(const void* gsrc, void* ldst, int bytes, int tid) {
    unsigned lbase = (unsigned)(uintptr_t)ldst;        // low 32 bits == LDS byte address
    const char* g = (const char*)gsrc;
    for (int off = tid * 16; off < bytes; off += 256 * 16) {
        unsigned laddr = lbase + (unsigned)off;
        const void* gaddr = (const void*)(g + off);
        asm volatile("global_load_async_to_lds_b128 %0, %1, off"
                     :: "v"(laddr), "v"(gaddr) : "memory");
    }
    asm volatile("s_wait_asynccnt 0" ::: "memory");
}

// positional table value: sub in [0,36): [0..7]=pe_y(P=10,F=4,idx=dy),
// [8..15]=pe_m(P=12,F=4,idx=mo), [16..27]=pe_d(P=31,F=6,idx=dd), [28..35]=pe_w(P=7,F=4,idx=dw)
__device__ __forceinline__ float pos_val(int sub, int dy, int mo, int dd, int dw) {
    float P; int F, idx, s0;
    if (sub < 8)       { P = 10.f; F = 4; idx = dy;  s0 = sub; }
    else if (sub < 16) { P = 12.f; F = 4; idx = mo;  s0 = sub - 8; }
    else if (sub < 28) { P = 31.f; F = 6; idx = dd;  s0 = sub - 16; }
    else               { P = 7.f;  F = 4; idx = dw;  s0 = sub - 28; }
    bool is_cos = (s0 >= F);
    int  f      = is_cos ? (s0 - F) : s0;
    float ang = (3.14159265358979323846f / P) * (float)(1 << f) * ((float)idx - 1.0f);
    return is_cos ? __cosf(ang) : __sinf(ang);
}

// ---- WMMA fragment gathers from LDS f16 staging buffers -------------------
__device__ __forceinline__ v16h load_fragA(const _Float16* p) {
    v8h a0 = *(const v8h*)(p);        // K = base .. base+7
    v8h a1 = *(const v8h*)(p + 16);   // K = base+16 .. base+23
    v16h out;
    #pragma unroll
    for (int i = 0; i < 8; ++i) { out[i] = a0[i]; out[i + 8] = a1[i]; }
    return out;
}
__device__ __forceinline__ v16h load_fragB(const _Float16* p) {
    v8h b0 = *(const v8h*)(p);
    v8h b1 = *(const v8h*)(p + 16);
    v16h out;
    #pragma unroll
    for (int i = 0; i < 8; ++i) { out[i] = b0[i]; out[i + 8] = b1[i]; }
    return out;
}

// Tiled GEMM over whole block: 8 waves round-robin on output 16x16 tiles.
// K-loop double-buffered so next ds_load_b128s are in flight across each WMMA.
template <typename Epi>
__device__ __forceinline__ void gemm_wmma(const _Float16* __restrict__ A,
                                          const _Float16* __restrict__ B,
                                          int Mtiles, int Ntiles, int Ktiles, int Kpad,
                                          int wave, int lane, Epi epi) {
    const int r  = lane & 15;
    const int hi = lane >> 4;
    const int nTiles = Mtiles * Ntiles;
    for (int tile = wave; tile < nTiles; tile += NWAVE) {
        const int mt = tile / Ntiles;
        const int nt = tile - mt * Ntiles;
        v8f acc = {};
        const _Float16* ap = A + (mt * 16 + r) * Kpad + hi * 8;
        const _Float16* bp = B + (nt * 16 + r) * Kpad + hi * 8;
        v16h av = load_fragA(ap);
        v16h bv = load_fragB(bp);
        #pragma unroll
        for (int kt = 0; kt < Ktiles - 1; ++kt) {
            ap += 32; bp += 32;
            v16h an = load_fragA(ap);
            v16h bn = load_fragB(bp);
            acc = __builtin_amdgcn_wmma_f32_16x16x32_f16(
                false, av, false, bv, (short)0, acc, false, false);
            av = an; bv = bn;
        }
        acc = __builtin_amdgcn_wmma_f32_16x16x32_f16(
            false, av, false, bv, (short)0, acc, false, false);
        epi(mt * 16, nt * 16 + r, hi, acc);
    }
}

// fallback: stage W (Kreal x Nreal f32) + bias row (at k==Kreal) into column-major f16
__device__ __forceinline__ void build_B(const float* __restrict__ W, const float* __restrict__ bias,
                                        int Kreal, int Nreal, int Kpad, int NcolsPad,
                                        _Float16* B16, int tid) {
    const int tot = NcolsPad * Kpad;
    for (int idx = tid; idx < tot; idx += 256) {
        int n = idx / Kpad, k = idx - n * Kpad;
        int nc = imini(n, Nreal - 1);
        float v = (k < Kreal) ? W[imini(k, Kreal - 1) * Nreal + nc]
                              : ((k == Kreal) ? bias[nc] : 0.f);
        v = (n < Nreal && k <= Kreal) ? v : 0.f;
        B16[idx] = (_Float16)v;
    }
}

__device__ __forceinline__ void zero_chunks(_Float16* p, int nchunks, int tid) {
    v8h z = {};
    for (int c = tid; c < nchunks; c += 256) *(v8h*)(p + c * 8) = z;
}

__device__ __forceinline__ void resblock(float* Xres, _Float16* A16, _Float16* B16, _Float16* H16,
    const _Float16* i_tw1, const _Float16* i_tw2, const _Float16* i_cw1, const _Float16* i_cw2,
    const float* tw1, const float* tb1, const float* tw2, const float* tb2,
    const float* cw1, const float* cb1, const float* cw2, const float* cb2,
    int tid, int wave, int lane) {
    const bool use_img = (i_tw1 != nullptr);
    // ---------------- time mixing: X += (relu(X^T tw1 + tb1) tw2 + tb2)^T ----
    for (int c = tid; c < 80 * 16; c += 256) {                 // A16 = X^T (80 rows x 128 K)
        int m = c >> 4, k0 = (c & 15) << 3;
        v8h o;
        #pragma unroll
        for (int i = 0; i < 8; ++i) {
            int k = k0 + i;
            float v = Xres[imini(k, XROWS - 1) * XS + m];
            float val = (m < DM && k < SEQL) ? v : 0.f;
            val = (k == SEQL) ? 1.f : val;                     // bias lane
            o[i] = (_Float16)val;
        }
        *(v8h*)(A16 + m * 128 + k0) = o;
    }
    if (use_img) async_stage(i_tw1, B16, IMG_TW1_SZ * 2, tid);
    else         build_B(tw1, tb1, SEQL, DM, 128, 80, B16, tid);
    zero_chunks(H16, 80 * 12, tid);                            // 80x96
    __syncthreads();
    gemm_wmma(A16, B16, 5, 5, 4, 128, wave, lane,              // K = 102 (incl. bias)
        [&](int mb, int n, int hi, const v8f& acc) {
            _Float16* hp = H16 + (mb + hi * 8) * 96 + n;
            #pragma unroll
            for (int rr = 0; rr < 8; ++rr)
                hp[rr * 96] = (_Float16)fmaxf(acc[rr], 0.f);
        });
    __syncthreads();
    if (use_img) async_stage(i_tw2, B16, IMG_TW2_SZ * 2, tid);
    else         build_B(tw2, tb2, DM, SEQL, 96, 112, B16, tid);
    if (tid < 80) H16[tid * 96 + DM] = (_Float16)1.f;          // bias lane of H
    __syncthreads();
    gemm_wmma(H16, B16, 5, 7, 3, 96, wave, lane,               // K = 73 (incl. bias)
        [&](int mb, int n, int hi, const v8f& acc) {           // transpose-add into X
            float* xp = Xres + n * XS + mb + hi * 8;
            #pragma unroll
            for (int rr = 0; rr < 8; ++rr)
                atomicAdd(xp + rr, acc[rr]);                   // ds_add_f32
        });
    __syncthreads();
    // ---------------- channel mixing: X += relu(X cw1 + cb1) cw2 + cb2 ------
    for (int c = tid; c < 112 * 9; c += 256) {                 // A16 = X, real K chunks
        int m = c / 9, k0 = (c - m * 9) << 3;                  // k0 in 0..64
        v4f x0 = *(const v4f*)(Xres + m * XS + k0);
        v4f x1 = *(const v4f*)(Xres + m * XS + k0 + 4);
        bool mr = (m < SEQL);
        v8h o;
        #pragma unroll
        for (int i = 0; i < 4; ++i) {
            o[i]     = (_Float16)((mr && k0 + i < DM)     ? x0[i] : 0.f);
            o[i + 4] = (_Float16)((mr && k0 + 4 + i < DM) ? x1[i] : 0.f);
        }
        *(v8h*)(A16 + m * 96 + k0) = o;
    }
    {   // K-tail chunks k0 in {72,80,88}: zero except bias lane 1.0 at k==72
        for (int c = tid; c < 112 * 3; c += 256) {
            int m = c / 3, t = c - m * 3;
            v8h z = {};
            if (t == 0) z[0] = (_Float16)1.f;
            *(v8h*)(A16 + m * 96 + 72 + (t << 3)) = z;
        }
    }
    if (use_img) async_stage(i_cw1, B16, IMG_CW_SZ * 2, tid);
    else         build_B(cw1, cb1, DM, DM, 96, 80, B16, tid);
    zero_chunks(H16, 112 * 12, tid);                           // 112x96
    __syncthreads();
    gemm_wmma(A16, B16, 7, 5, 3, 96, wave, lane,               // K = 73 (incl. bias)
        [&](int mb, int n, int hi, const v8f& acc) {
            _Float16* hp = H16 + (mb + hi * 8) * 96 + n;
            #pragma unroll
            for (int rr = 0; rr < 8; ++rr)
                hp[rr * 96] = (_Float16)fmaxf(acc[rr], 0.f);
        });
    __syncthreads();
    if (use_img) async_stage(i_cw2, B16, IMG_CW_SZ * 2, tid);
    else         build_B(cw2, cb2, DM, DM, 96, 80, B16, tid);
    if (tid < 112) H16[tid * 96 + DM] = (_Float16)1.f;         // bias lane of R
    __syncthreads();
    gemm_wmma(H16, B16, 7, 5, 3, 96, wave, lane,               // K = 73 (incl. bias)
        [&](int mb, int n, int hi, const v8f& acc) {
            float* xp = Xres + (mb + hi * 8) * XS + n;
            #pragma unroll
            for (int rr = 0; rr < 8; ++rr)
                atomicAdd(xp + rr * XS, acc[rr]);
        });
    __syncthreads();
}

// one-time weight image pre-pass: f32 (Kreal x Nreal) + bias -> padded col-major f16
__global__ void __launch_bounds__(256)
weight_prep_kernel(const float* __restrict__ W, const float* __restrict__ bias,
                   int Kreal, int Nreal, int Kpad, int NcolsPad, _Float16* __restrict__ dst) {
    int idx = blockIdx.x * 256 + threadIdx.x;
    int tot = NcolsPad * Kpad;
    if (idx >= tot) return;
    int n = idx / Kpad, k = idx - n * Kpad;
    int nc = imini(n, Nreal - 1);
    float v = (k < Kreal) ? W[imini(k, Kreal - 1) * Nreal + nc]
                          : ((k == Kreal) ? bias[nc] : 0.f);
    v = (n < Nreal && k <= Kreal) ? v : 0.f;
    dst[idx] = (_Float16)v;
}

__global__ void __launch_bounds__(256)
forecastpfn_tsmixer_kernel(
    const float* __restrict__ history, const int* __restrict__ ts,
    const int* __restrict__ target_ts, const int* __restrict__ task,
    const float* __restrict__ w_np, const float* __restrict__ b_np,
    const float* __restrict__ w_wp, const float* __restrict__ b_wp,
    const float* __restrict__ task_emb,
    const float* tw1_0, const float* tb1_0, const float* tw2_0, const float* tb2_0,
    const float* cw1_0, const float* cb1_0, const float* cw2_0, const float* cb2_0,
    const float* tw1_1, const float* tb1_1, const float* tw2_1, const float* tb2_1,
    const float* cw1_1, const float* cb1_1, const float* cw2_1, const float* cb2_1,
    const float* __restrict__ wp_time, const float* __restrict__ bp_time,
    const float* __restrict__ w_head, const float* __restrict__ b_head,
    const _Float16* __restrict__ wimg,
    float* __restrict__ out, int batch) {
    extern __shared__ __align__(16) char smem[];
    float*    Xres = (float*)smem;
    _Float16* A16  = (_Float16*)(smem + XRES_BYTES);
    _Float16* B16  = (_Float16*)(smem + XRES_BYTES + AB_BYTES);
    _Float16* H16  = (_Float16*)(smem + XRES_BYTES + 2 * AB_BYTES);
    float*    scr  = (float*)(smem + XRES_BYTES + 3 * AB_BYTES);
    float*    vbuf = scr + 8;     // 72
    float*    hbuf = scr + 80;    // 100

    const int b    = blockIdx.x;
    const int tid  = threadIdx.x;
    const int wave = tid >> 5;
    const int lane = tid & 31;

    // stage history row into LDS once (async copy, 25 x b128)
    async_stage(history + (size_t)b * T_HIST, hbuf, T_HIST * 4, tid);
    __syncthreads();

    // ---------------- robust_scale (wave 0, shuffle reductions) -------------
    if (wave == 0) {
        float s_sum = 0.f, s_cnt = 0.f, s_all = 0.f;
        for (int t = lane; t < T_HIST; t += 32) {
            float hv = hbuf[t];
            s_all += hv;
            if (hv != 0.f) { s_sum += hv; s_cnt += 1.f; }
        }
        for (int o = 16; o > 0; o >>= 1) {
            s_sum += __shfl_xor(s_sum, o, 32);
            s_cnt += __shfl_xor(s_cnt, o, 32);
            s_all += __shfl_xor(s_all, o, 32);
        }
        float mean = (s_cnt > 0.f) ? s_sum / fmaxf(s_cnt, 1.f) : 0.f;
        float s_var = 0.f;
        for (int t = lane; t < T_HIST; t += 32) {
            float hv = hbuf[t];
            if (hv != 0.f) { float d = hv - mean; s_var += d * d; }
        }
        for (int o = 16; o > 0; o >>= 1) s_var += __shfl_xor(s_var, o, 32);
        float stdv  = (s_cnt > 0.f) ? sqrtf(s_var / fmaxf(s_cnt, 1.f)) : 0.f;
        float upper = mean + 2.f * stdv;
        float c_sum = 0.f, c_cnt = 0.f;
        for (int t = lane; t < T_HIST; t += 32) {
            float cv = fminf(fmaxf(hbuf[t], 0.f), upper);
            if (cv != 0.f) { c_sum += cv; c_cnt += 1.f; }
        }
        for (int o = 16; o > 0; o >>= 1) {
            c_sum += __shfl_xor(c_sum, o, 32);
            c_cnt += __shfl_xor(c_cnt, o, 32);
        }
        float mc = (c_cnt > 0.f) ? c_sum / fmaxf(c_cnt, 1.f) : 0.f;
        float c_var = 0.f;
        for (int t = lane; t < T_HIST; t += 32) {
            float cv = fminf(fmaxf(hbuf[t], 0.f), upper);
            if (cv != 0.f) { float d = cv - mc; c_var += d * d; }
        }
        for (int o = 16; o > 0; o >>= 1) c_var += __shfl_xor(c_var, o, 32);
        float sc = (c_cnt > 0.f) ? sqrtf(c_var / fmaxf(c_cnt, 1.f)) : 0.f;
        if (lane == 0) {
            scr[0] = mc + sc + 1e-4f;        // s
            scr[1] = s_all / (float)T_HIST;  // raw history mean
        }
    }
    __syncthreads();
    const float s_scale = scr[0];
    const float hmean   = scr[1];
    const int   yr_last = ts[((size_t)b * T_HIST + (T_HIST - 1)) * 4 + 0];

    // ---------------- build padded X (112 x 80, real 101 x 72) --------------
    for (int idx = tid; idx < XROWS * XS; idx += 256) {
        int t = idx / XS, j = idx - t * XS;
        float val = 0.f;
        if (t < SEQL && j < DM) {
            if (t < T_HIST) {
                float sh = fminf(fmaxf(hbuf[t] / s_scale, 0.f), 3.f);
                if (j < 36) {
                    val = fmaxf(sh * w_np[j] + b_np[j], 0.f);
                } else {
                    int jj = j - 36;
                    const int* tp = ts + ((size_t)b * T_HIST + t) * 4;
                    int dy = iclampi(yr_last - tp[0], 0, 10);
                    val = fmaxf(sh * w_wp[jj] + b_wp[jj], 0.f) +
                          pos_val(jj, dy, tp[1], tp[2], tp[3]);
                }
            } else {                                            // target row (t == 100)
                int jm = (j < 36) ? j : j - 36;
                float te = task_emb[task[b] * 36 + jm];
                if (j < 36) {
                    val = te;
                } else {
                    const int* tt = target_ts + (size_t)b * 5;
                    int dy = iclampi(yr_last - tt[0], 0, 10);
                    val = te + pos_val(j - 36, dy, tt[1], tt[2], tt[3]);
                }
            }
        }
        Xres[idx] = val;
    }
    __syncthreads();

    resblock(Xres, A16, B16, H16,
             wimg ? wimg + OFF_TW1_0 : nullptr, wimg ? wimg + OFF_TW2_0 : nullptr,
             wimg ? wimg + OFF_CW1_0 : nullptr, wimg ? wimg + OFF_CW2_0 : nullptr,
             tw1_0, tb1_0, tw2_0, tb2_0, cw1_0, cb1_0, cw2_0, cb2_0, tid, wave, lane);
    resblock(Xres, A16, B16, H16,
             wimg ? wimg + OFF_TW1_1 : nullptr, wimg ? wimg + OFF_TW2_1 : nullptr,
             wimg ? wimg + OFF_CW1_1 : nullptr, wimg ? wimg + OFF_CW2_1 : nullptr,
             tw1_1, tb1_1, tw2_1, tb2_1, cw1_1, cb1_1, cw2_1, cb2_1, tid, wave, lane);

    // ---------------- head: (101->1) over time, (72->1) over channels -------
    if (tid < DM) {
        float acc = bp_time[0];
        for (int t = 0; t < SEQL; ++t) acc += Xres[t * XS + tid] * wp_time[t];
        vbuf[tid] = acc;
    }
    __syncthreads();
    if (tid == 0) {
        float y = b_head[0];
        for (int c = 0; c < DM; ++c) y += vbuf[c] * w_head[c];
        y = fmaxf(y, 0.f);
        out[b]         = y * s_scale + hmean;   // y output
        out[batch + b] = s_scale;               // scale output (B,1,1)
    }
}

extern "C" void kernel_launch(void* const* d_in, const int* in_sizes, int n_in,
                              void* d_out, int out_size, void* d_ws, size_t ws_size,
                              hipStream_t stream) {
    (void)n_in; (void)out_size;
    const float* history   = (const float*)d_in[0];
    const int*   ts        = (const int*)d_in[1];
    const int*   target_ts = (const int*)d_in[2];
    const int*   task      = (const int*)d_in[3];
    const float* w_np   = (const float*)d_in[4];
    const float* b_np   = (const float*)d_in[5];
    const float* w_wp   = (const float*)d_in[6];
    const float* b_wp   = (const float*)d_in[7];
    const float* temb   = (const float*)d_in[8];
    const float* tw1_0  = (const float*)d_in[9];
    const float* tb1_0  = (const float*)d_in[10];
    const float* tw2_0  = (const float*)d_in[11];
    const float* tb2_0  = (const float*)d_in[12];
    const float* cw1_0  = (const float*)d_in[13];
    const float* cb1_0  = (const float*)d_in[14];
    const float* cw2_0  = (const float*)d_in[15];
    const float* cb2_0  = (const float*)d_in[16];
    const float* tw1_1  = (const float*)d_in[17];
    const float* tb1_1  = (const float*)d_in[18];
    const float* tw2_1  = (const float*)d_in[19];
    const float* tb2_1  = (const float*)d_in[20];
    const float* cw1_1  = (const float*)d_in[21];
    const float* cb1_1  = (const float*)d_in[22];
    const float* cw2_1  = (const float*)d_in[23];
    const float* cb2_1  = (const float*)d_in[24];
    const float* wp_time = (const float*)d_in[25];
    const float* bp_time = (const float*)d_in[26];
    const float* w_head  = (const float*)d_in[27];
    const float* b_head  = (const float*)d_in[28];

    const int batch = in_sizes[3];   // task has B elements

    _Float16* wimg = nullptr;
    if (ws_size >= WS_NEEDED) {
        wimg = (_Float16*)d_ws;
        struct { const float* W; const float* bias; int Kr, Nr, Kp, Nc; int off; } jobs[8] = {
            { tw1_0, tb1_0, SEQL, DM,   128,  80, OFF_TW1_0 },
            { tw2_0, tb2_0, DM,   SEQL,  96, 112, OFF_TW2_0 },
            { cw1_0, cb1_0, DM,   DM,    96,  80, OFF_CW1_0 },
            { cw2_0, cb2_0, DM,   DM,    96,  80, OFF_CW2_0 },
            { tw1_1, tb1_1, SEQL, DM,   128,  80, OFF_TW1_1 },
            { tw2_1, tb2_1, DM,   SEQL,  96, 112, OFF_TW2_1 },
            { cw1_1, cb1_1, DM,   DM,    96,  80, OFF_CW1_1 },
            { cw2_1, cb2_1, DM,   DM,    96,  80, OFF_CW2_1 },
        };
        for (int i = 0; i < 8; ++i) {
            int tot = jobs[i].Nc * jobs[i].Kp;
            weight_prep_kernel<<<dim3((tot + 255) / 256), dim3(256), 0, stream>>>(
                jobs[i].W, jobs[i].bias, jobs[i].Kr, jobs[i].Nr, jobs[i].Kp, jobs[i].Nc,
                wimg + jobs[i].off);
        }
    }

    (void)hipFuncSetAttribute((const void*)forecastpfn_tsmixer_kernel,
                              hipFuncAttributeMaxDynamicSharedMemorySize, SMEM_BYTES);

    forecastpfn_tsmixer_kernel<<<dim3(batch), dim3(256), SMEM_BYTES, stream>>>(
        history, ts, target_ts, task, w_np, b_np, w_wp, b_wp, temb,
        tw1_0, tb1_0, tw2_0, tb2_0, cw1_0, cb1_0, cw2_0, cb2_0,
        tw1_1, tb1_1, tw2_1, tb2_1, cw1_1, cb1_1, cw2_1, cb2_1,
        wp_time, bp_time, w_head, b_head, wimg, (float*)d_out, batch);
}
